// CrossAttn_23888608100978
// MI455X (gfx1250) — matrix operand order repaired and checked by reading, not compiled, not run on hardware
//
#include <hip/hip_runtime.h>
#include <hip/hip_bf16.h>
#include <math.h>

// ---------------- problem constants ----------------
#define D_DIM   512
#define C_DIM   4096
#define S_SUP   50000
#define N_TOT   100000
#define BM      128              // support rows per workgroup
#define NBLK    391              // ceil(S_SUP / BM)
#define S_PAD   (NBLK * BM)      // 50048 (padded, zero rows at tail)
#define NCHUNK  32               // codebook columns staged per LDS chunk
#define KSTEPS  16               // 512 / 32
#define BSTRIDE 520              // bf16 elems per LDS row (8-elem pad vs bank conflicts)

typedef __bf16 bf16_t;
typedef bf16_t v16bf __attribute__((ext_vector_type(16)));
typedef bf16_t v8bf  __attribute__((ext_vector_type(8)));
typedef float  v8f   __attribute__((ext_vector_type(8)));
typedef int    v4i_b128 __attribute__((vector_size(16)));   // matches builtin param type

union Frag16 { v16bf v; v8bf h[2]; };

#define GLOBAL_AS __attribute__((address_space(1)))
#define LDS_AS    __attribute__((address_space(3)))

// ---- gfx1250 async global->LDS path (guarded; fallback = register staging) ----
#if defined(__has_builtin)
#  if __has_builtin(__builtin_amdgcn_global_load_async_to_lds_b128)
#    define ASYNC_LDS 1
#  endif
#endif
#ifndef ASYNC_LDS
#  define ASYNC_LDS 0
#endif

#if ASYNC_LDS
#  if __has_builtin(__builtin_amdgcn_s_wait_asynccnt)
#    define WAIT_ASYNC(n) __builtin_amdgcn_s_wait_asynccnt(n)
#  else
#    define WAIT_ASYNC(n) asm volatile("s_wait_asynccnt %0" :: "n"(n) : "memory")
#  endif
#endif

// ---------------- prep kernels ----------------
__global__ void zero_kernel(unsigned int* counts, float* proto_sum, float* pc_sum) {
    int i = blockIdx.x * 256 + threadIdx.x;
    if (i < C_DIM) counts[i] = 0u;
    if (i < D_DIM) { proto_sum[i] = 0.0f; pc_sum[i] = 0.0f; }
}

// support rows fp32 -> bf16, padded to S_PAD rows (pad rows = 0)
__global__ void convertA_kernel(const float* __restrict__ X, bf16_t* __restrict__ Abf) {
    size_t total = (size_t)S_PAD * D_DIM;
    for (size_t i = (size_t)blockIdx.x * 256 + threadIdx.x; i < total;
         i += (size_t)gridDim.x * 256) {
        size_t row = i >> 9;                       // i / 512
        float v = (row < (size_t)S_SUP) ? X[i] : 0.0f;
        Abf[i] = (bf16_t)v;
    }
}

__global__ void convertB_kernel(const float* __restrict__ CB, bf16_t* __restrict__ Bbf) {
    size_t total = (size_t)C_DIM * D_DIM;
    for (size_t i = (size_t)blockIdx.x * 256 + threadIdx.x; i < total;
         i += (size_t)gridDim.x * 256) {
        Bbf[i] = (bf16_t)CB[i];
    }
}

// ---------------- core: WMMA GEMM + running argmax -> class counts ----------------
__global__ __launch_bounds__(256)
void gemm_argmax_kernel(const bf16_t* __restrict__ Abf,
                        const bf16_t* __restrict__ Bbf,
                        unsigned int* __restrict__ counts) {
#if ASYNC_LDS
    __shared__ bf16_t Blds[2 * NCHUNK * BSTRIDE];   // double buffer, 2*33280 B
#else
    __shared__ bf16_t Blds[NCHUNK * BSTRIDE];       // single buffer, 33280 B
#endif

    const int tid  = threadIdx.x;
    const int wave = tid >> 5;
    const int lane = tid & 31;
    const int half = lane >> 4;
    const int l15  = lane & 15;

    const int rowBlock = blockIdx.x * BM;
    const int myRow    = rowBlock + wave * 16 + l15;   // A row this lane carries

    // --- preload this wave's A slice (16 rows x 512 K) into registers ---
    // A 16-bit 16x32 layout: lanes 0-15 K{0..7,16..23}, lanes 16-31 K{8..15,24..31}
    Frag16 areg[KSTEPS];
    {
        const bf16_t* arow = Abf + (size_t)myRow * D_DIM;
        #pragma unroll
        for (int ks = 0; ks < KSTEPS; ++ks) {
            const int kb = ks * 32 + half * 8;
            areg[ks].h[0] = *(const v8bf*)(arow + kb);        // K kb..kb+7
            areg[ks].h[1] = *(const v8bf*)(arow + kb + 16);   // K kb+16..kb+23
        }
    }

    float best[8];
    int   bidx[8];
    #pragma unroll
    for (int r = 0; r < 8; ++r) { best[r] = -3.4e38f; bidx[r] = 0; }

    // per-thread staging geometry: fixed 16B column, 8 rows (stride 4) per stage
    const int ccc = (tid & 63) * 8;   // bf16 element offset within a row
    const int rr0 = tid >> 6;         // 0..3

    // compute one staged chunk (2 N-tiles x 16 K-steps), software-pipelined B frags
    auto computeChunk = [&](int c0, int bufSel) {
        const bf16_t* base = &Blds[bufSel * (NCHUNK * BSTRIDE)];
        const bf16_t* b0p  = base + l15 * BSTRIDE + half * 8;
        const bf16_t* b1p  = b0p + 16 * BSTRIDE;
        v8f acc0 = {0.f,0.f,0.f,0.f,0.f,0.f,0.f,0.f};
        v8f acc1 = {0.f,0.f,0.f,0.f,0.f,0.f,0.f,0.f};
        Frag16 c0f, c1f;
        c0f.h[0] = *(const v8bf*)(b0p);     c0f.h[1] = *(const v8bf*)(b0p + 16);
        c1f.h[0] = *(const v8bf*)(b1p);     c1f.h[1] = *(const v8bf*)(b1p + 16);
        #pragma unroll
        for (int ks = 0; ks < KSTEPS; ++ks) {
            Frag16 n0f, n1f;
            if (ks + 1 < KSTEPS) {          // prefetch next K-step's B fragments
                const int o = (ks + 1) * 32;
                n0f.h[0] = *(const v8bf*)(b0p + o);
                n0f.h[1] = *(const v8bf*)(b0p + o + 16);
                n1f.h[0] = *(const v8bf*)(b1p + o);
                n1f.h[1] = *(const v8bf*)(b1p + o + 16);
            }
            acc0 = __builtin_amdgcn_wmma_f32_16x16x32_bf16(
                     false, areg[ks].v, false, c0f.v, (short)0, acc0, false, false);
            acc1 = __builtin_amdgcn_wmma_f32_16x16x32_bf16(
                     false, areg[ks].v, false, c1f.v, (short)0, acc1, false, false);
            if (ks + 1 < KSTEPS) { c0f = n0f; c1f = n1f; }
        }
        const int col0 = c0 + l15;
        #pragma unroll
        for (int r = 0; r < 8; ++r) {
            float v0 = acc0[r];             // element (m = half*8 + r, n = l15)
            if (v0 > best[r]) { best[r] = v0; bidx[r] = col0; }
            float v1 = acc1[r];
            if (v1 > best[r]) { best[r] = v1; bidx[r] = col0 + 16; }
        }
    };

#if ASYNC_LDS
    // async DMA staging: global -> LDS directly, double-buffered
    auto stage = [&](int c0, int bufSel) {
        #pragma unroll
        for (int it = 0; it < 8; ++it) {
            const int rr = rr0 + it * 4;
            GLOBAL_AS v4i_b128* g =
                (GLOBAL_AS v4i_b128*)(Bbf + (size_t)(c0 + rr) * D_DIM + ccc);
            LDS_AS v4i_b128* l =
                (LDS_AS v4i_b128*)(&Blds[bufSel * (NCHUNK * BSTRIDE) + rr * BSTRIDE + ccc]);
            __builtin_amdgcn_global_load_async_to_lds_b128(g, l, 0, 0);
        }
    };

    stage(0, 0);
    int c0 = 0;
    const int NC = C_DIM / NCHUNK;
    for (int c = 0; c < NC - 1; ++c, c0 += NCHUNK) {
        stage(c0 + NCHUNK, (c + 1) & 1);   // kick off next chunk's DMA
        WAIT_ASYNC(8);                      // this wave's chunk-c transfers done
        __syncthreads();                    // everyone's transfers visible
        computeChunk(c0, c & 1);
        __syncthreads();                    // chunk buffer free for reuse
    }
    WAIT_ASYNC(0);
    __syncthreads();
    computeChunk(C_DIM - NCHUNK, (NC - 1) & 1);
#else
    // fallback: batched register staging with one-chunk global prefetch
    v8bf g[8];
    #pragma unroll
    for (int it = 0; it < 8; ++it)
        g[it] = *(const v8bf*)(Bbf + (size_t)(rr0 + it * 4) * D_DIM + ccc);

    for (int c0 = 0; c0 < C_DIM; c0 += NCHUNK) {
        #pragma unroll
        for (int it = 0; it < 8; ++it)
            *(v8bf*)(&Blds[(rr0 + it * 4) * BSTRIDE + ccc]) = g[it];
        __syncthreads();
        if (c0 + NCHUNK < C_DIM) {         // prefetch next chunk while computing
            #pragma unroll
            for (int it = 0; it < 8; ++it)
                g[it] = *(const v8bf*)(Bbf +
                        (size_t)(c0 + NCHUNK + rr0 + it * 4) * D_DIM + ccc);
        }
        computeChunk(c0, 0);
        __syncthreads();
    }
#endif

    // --- merge argmax across the 16 lanes (N positions) holding each row ---
    #pragma unroll
    for (int off = 8; off >= 1; off >>= 1) {
        #pragma unroll
        for (int r = 0; r < 8; ++r) {
            float ov = __shfl_xor(best[r], off, 32);
            int   oi = __shfl_xor(bidx[r], off, 32);
            if (ov > best[r] || (ov == best[r] && oi < bidx[r])) {
                best[r] = ov; bidx[r] = oi;
            }
        }
    }
    if (l15 == 0) {
        #pragma unroll
        for (int r = 0; r < 8; ++r) {
            const int m    = half * 8 + r;
            const int grow = rowBlock + wave * 16 + m;
            if (grow < S_SUP) atomicAdd(&counts[bidx[r]], 1u);
        }
    }
}

// ---------------- proto = mean(support) : column sums ----------------
#define ROWS_PER_CS 2000
__global__ void colsum_kernel(const float* __restrict__ X, float* __restrict__ proto_sum) {
    const int d0 = threadIdx.x;                  // 0..255
    const int r0 = blockIdx.x * ROWS_PER_CS;
    float a0 = 0.f, a1 = 0.f;
    for (int r = r0; r < r0 + ROWS_PER_CS; ++r) {
        const float* row = X + (size_t)r * D_DIM;
        a0 += row[d0];
        a1 += row[d0 + 256];
    }
    atomicAdd(&proto_sum[d0],       a0);
    atomicAdd(&proto_sum[d0 + 256], a1);
}

// ---------------- proto_code = (1/S) * sum_c counts[c] * codebook[c] ----------------
__global__ void codesum_kernel(const float* __restrict__ CB,
                               const unsigned int* __restrict__ counts,
                               float* __restrict__ pc_sum) {
    const int d0 = threadIdx.x;
    const int c0 = blockIdx.x * 256;             // 16 blocks
    float a0 = 0.f, a1 = 0.f;
    for (int c = c0; c < c0 + 256; ++c) {
        const float w = (float)counts[c];
        const float* row = CB + (size_t)c * D_DIM;
        a0 += w * row[d0];
        a1 += w * row[d0 + 256];
    }
    atomicAdd(&pc_sum[d0],       a0);
    atomicAdd(&pc_sum[d0 + 256], a1);
}

// ---------------- final: per-query distances ----------------
__global__ __launch_bounds__(256)
void final_kernel(const float* __restrict__ X,
                  const float* __restrict__ proto_sum,
                  const float* __restrict__ pc_sum,
                  float* __restrict__ out) {
    __shared__ float p[D_DIM];
    __shared__ float pc[D_DIM];
    const float invS = 1.0f / (float)S_SUP;
    for (int i = threadIdx.x; i < D_DIM; i += 256) {
        p[i]  = proto_sum[i] * invS;
        pc[i] = pc_sum[i]    * invS;
    }
    __syncthreads();

    const int lane = threadIdx.x & 31;
    const int wave = threadIdx.x >> 5;
    const int wavesTotal = gridDim.x * 8;
    const int nq = N_TOT - S_SUP;                // 50000 queries
    for (int row = blockIdx.x * 8 + wave; row < nq; row += wavesTotal) {
        const float* q = X + (size_t)(S_SUP + row) * D_DIM;
        float a = 0.f, b = 0.f;
        for (int d = lane; d < D_DIM; d += 32) {
            const float qv = q[d];
            const float dx = qv - p[d];  a += dx * dx;
            const float dy = qv - pc[d]; b += dy * dy;
        }
        #pragma unroll
        for (int off = 16; off >= 1; off >>= 1) {
            a += __shfl_xor(a, off, 32);
            b += __shfl_xor(b, off, 32);
        }
        if (lane == 0) out[row] = 0.5f * (sqrtf(a) + sqrtf(b));
    }
}

// ---------------- host side ----------------
extern "C" void kernel_launch(void* const* d_in, const int* in_sizes, int n_in,
                              void* d_out, int out_size, void* d_ws, size_t ws_size,
                              hipStream_t stream) {
    const float* X  = (const float*)d_in[0];       // (N, D) fp32
    const float* CB = (const float*)d_in[1];       // (C, D) fp32
    float* out = (float*)d_out;                    // (N - S) fp32

    // workspace layout (bytes)
    char* ws = (char*)d_ws;
    size_t off = 0;
    bf16_t* Abf = (bf16_t*)(ws + off); off += (size_t)S_PAD * D_DIM * sizeof(bf16_t);
    bf16_t* Bbf = (bf16_t*)(ws + off); off += (size_t)C_DIM * D_DIM * sizeof(bf16_t);
    off = (off + 255) & ~(size_t)255;
    unsigned int* counts = (unsigned int*)(ws + off); off += C_DIM * sizeof(unsigned int);
    float* proto_sum = (float*)(ws + off); off += D_DIM * sizeof(float);
    float* pc_sum    = (float*)(ws + off); off += D_DIM * sizeof(float);

    zero_kernel<<<16, 256, 0, stream>>>(counts, proto_sum, pc_sum);
    convertA_kernel<<<4096, 256, 0, stream>>>(X, Abf);
    convertB_kernel<<<1024, 256, 0, stream>>>(CB, Bbf);
    gemm_argmax_kernel<<<NBLK, 256, 0, stream>>>(Abf, Bbf, counts);
    colsum_kernel<<<S_SUP / ROWS_PER_CS, 256, 0, stream>>>(X, proto_sum);
    codesum_kernel<<<C_DIM / 256, 256, 0, stream>>>(CB, counts, pc_sum);
    final_kernel<<<512, 256, 0, stream>>>(X, proto_sum, pc_sum, out);
}
